// mLSTM_57312043597977
// MI455X (gfx1250) — matrix operand order repaired and checked by reading, not compiled, not run on hardware
//
#include <hip/hip_runtime.h>

typedef _Float16 h16;
typedef __attribute__((ext_vector_type(8)))  _Float16 v8h;
typedef __attribute__((ext_vector_type(16))) _Float16 v16h;
typedef __attribute__((ext_vector_type(8)))  float    v8f;

#define B_SZ   256
#define I_SZ   1024
#define H_SZ   8
#define D_SZ   128
#define P_SZ   2048
#define HD_SZ  1024
#define KCONV  4
#define EPSV   1e-5f
#define INV_SQRT_D 0.08838834764831845f

#if defined(__gfx1250__) && __has_builtin(__builtin_amdgcn_global_load_async_to_lds_b32) && __has_builtin(__builtin_amdgcn_s_wait_asynccnt)
#define ASYNC_LDS 1
#define GPTR(p) ((__attribute__((address_space(1))) int*)(p))
#define LPTR(p) ((__attribute__((address_space(3))) int*)(p))
#else
#define ASYNC_LDS 0
#endif

__device__ __forceinline__ float sigmoidf_(float x){ return 1.0f/(1.0f+__expf(-x)); }
__device__ __forceinline__ float siluf_(float x){ return x*sigmoidf_(x); }

// ---------------------------------------------------------------------------
// f32 -> f16 conversion (grid-stride)
// ---------------------------------------------------------------------------
__global__ void f32_to_h16_kernel(const float* __restrict__ s, h16* __restrict__ d, int n){
  for (int i = blockIdx.x*blockDim.x + threadIdx.x; i < n; i += gridDim.x*blockDim.x)
    d[i] = (h16)s[i];
}

// ---------------------------------------------------------------------------
// LayerNorm over I=1024, writes f16 activations for the WMMA GEMMs
// ---------------------------------------------------------------------------
__global__ __launch_bounds__(256)
void layernorm_kernel(const float* __restrict__ x, const float* __restrict__ g,
                      const float* __restrict__ b, h16* __restrict__ xn){
  __shared__ float red[256];
  const int row = blockIdx.x, tid = threadIdx.x;
  const float4 v = ((const float4*)(x + (size_t)row*I_SZ))[tid];
  float s  = v.x+v.y+v.z+v.w;
  float ss = v.x*v.x+v.y*v.y+v.z*v.z+v.w*v.w;
  red[tid]=s; __syncthreads();
  for (int st=128;st>0;st>>=1){ if(tid<st) red[tid]+=red[tid+st]; __syncthreads(); }
  const float mu = red[0]*(1.0f/I_SZ); __syncthreads();
  red[tid]=ss; __syncthreads();
  for (int st=128;st>0;st>>=1){ if(tid<st) red[tid]+=red[tid+st]; __syncthreads(); }
  const float var = red[0]*(1.0f/I_SZ) - mu*mu;
  const float inv = rsqrtf(var + EPSV);
  const int j = tid*4;
  h16* o = xn + (size_t)row*I_SZ + j;
  o[0]=(h16)((v.x-mu)*inv*g[j+0]+b[j+0]);
  o[1]=(h16)((v.y-mu)*inv*g[j+1]+b[j+1]);
  o[2]=(h16)((v.z-mu)*inv*g[j+2]+b[j+2]);
  o[3]=(h16)((v.w-mu)*inv*g[j+3]+b[j+3]);
}

// ---------------------------------------------------------------------------
// WMMA GEMM: C[M,N] = A[M,K] @ W[N,K]^T (+bias[N]) (+addend[M,N])
// One wave per block (32 threads), 32x32 output tile = 2x2 v_wmma tiles.
// No LDS / no inter-wave sharing -> single-wave blocks maximize spread
// across WGPs (512 blocks for the N=2048 GEMMs).
// A frag per ISA 16-bit A layout: lane r=l&15 -> row, hi=l>>4 K-phase,
// two contiguous 16B runs (k0+hi*8, k0+16+hi*8). B frag: lane -> weight row
// (output column), same K order: two 16B runs at k0+hi*8, k0+16+hi*8.
// ---------------------------------------------------------------------------
__device__ __forceinline__ v16h load_frag_a(const h16* __restrict__ p, int k0){
  const v8h lo = *(const v8h*)(p + k0);
  const v8h hh = *(const v8h*)(p + k0 + 16);
  v16h f;
#pragma unroll
  for (int i=0;i<8;++i){ f[i]=lo[i]; f[i+8]=hh[i]; }
  return f;
}
__device__ __forceinline__ v16h load_frag_b(const h16* __restrict__ p, int k0){
  const v8h lo = *(const v8h*)(p + k0);
  const v8h hh = *(const v8h*)(p + k0 + 16);
  v16h f;
#pragma unroll
  for (int i=0;i<8;++i){ f[i]=lo[i]; f[i+8]=hh[i]; }
  return f;
}

__global__ __launch_bounds__(32)
void wmma_gemm_kernel(const h16* __restrict__ A, const h16* __restrict__ W,
                      const float* __restrict__ bias, const float* __restrict__ addend,
                      float* __restrict__ C, int M, int N, int K){
  const int lane = threadIdx.x;
  const int row0 = blockIdx.y*32;
  const int col0 = blockIdx.x*32;
  const int r = lane & 15, hi = lane >> 4;
  const h16* pa0 = A + (size_t)(row0 +      r)*K + hi*8;
  const h16* pa1 = A + (size_t)(row0 + 16 + r)*K + hi*8;
  const h16* pb0 = W + (size_t)(col0 +      r)*K + hi*8;
  const h16* pb1 = W + (size_t)(col0 + 16 + r)*K + hi*8;

  v8f acc00 = {}, acc01 = {}, acc10 = {}, acc11 = {};
  for (int k0 = 0; k0 < K; k0 += 32){
    // stream-prefetch the weight rows ~1KB ahead (gfx1250 global_prefetch_b8)
    __builtin_prefetch(pb0 + k0 + 512, 0, 1);
    __builtin_prefetch(pb1 + k0 + 512, 0, 1);
    const v16h a0 = load_frag_a(pa0, k0);
    const v16h a1 = load_frag_a(pa1, k0);
    const v16h b0 = load_frag_b(pb0, k0);
    const v16h b1 = load_frag_b(pb1, k0);
    acc00 = __builtin_amdgcn_wmma_f32_16x16x32_f16(false, a0, false, b0, (short)0, acc00, false, false);
    acc01 = __builtin_amdgcn_wmma_f32_16x16x32_f16(false, a0, false, b1, (short)0, acc01, false, false);
    acc10 = __builtin_amdgcn_wmma_f32_16x16x32_f16(false, a1, false, b0, (short)0, acc10, false, false);
    acc11 = __builtin_amdgcn_wmma_f32_16x16x32_f16(false, a1, false, b1, (short)0, acc11, false, false);
  }
  v8f accs[2][2] = {{acc00, acc01}, {acc10, acc11}};
#pragma unroll
  for (int mi=0; mi<2; ++mi){
#pragma unroll
    for (int ni=0; ni<2; ++ni){
      const v8f a = accs[mi][ni];
      const int col = col0 + ni*16 + r;
      const float bv = bias ? bias[col] : 0.0f;
#pragma unroll
      for (int j=0;j<8;++j){
        const int rw = row0 + mi*16 + hi*8 + j;       // C layout: M = vgpr + 8*hi
        const size_t idx = (size_t)rw * N + col;
        float v = a[j] + bv;
        if (addend) v += addend[idx];
        C[idx] = v;
      }
    }
  }
}

// ---------------------------------------------------------------------------
// Causal conv (K=4) along feature axis + SiLU; also emit f16 copies
// ---------------------------------------------------------------------------
__global__ void conv_silu_kernel(const float* __restrict__ xt, const float* __restrict__ cw,
                                 const float* __restrict__ cb, float* __restrict__ xc,
                                 h16* __restrict__ xc16, h16* __restrict__ xt16){
  const int g = blockIdx.x*blockDim.x + threadIdx.x;
  if (g >= B_SZ*P_SZ) return;
  const int p = g & (P_SZ-1);
  float acc = cb[0];
#pragma unroll
  for (int j=0;j<KCONV;++j){
    const int idx = p + j - (KCONV-1);
    if (idx >= 0) acc += cw[j]*xt[g + j - (KCONV-1)];
  }
  const float v = siluf_(acc);
  xc[g] = v; xc16[g] = (h16)v; xt16[g] = (h16)xt[g];
}

// ---------------------------------------------------------------------------
// Tiny projections i_t,f_t (N=8 each): one block per batch, one wave/output
// ---------------------------------------------------------------------------
__global__ __launch_bounds__(256)
void ifgate_kernel(const float* __restrict__ xc,
                   const float* __restrict__ Wi, const float* __restrict__ bi,
                   const float* __restrict__ Wf, const float* __restrict__ bf,
                   float* __restrict__ it, float* __restrict__ ft){
  const int b = blockIdx.x;
  const int lane = threadIdx.x & 31;
  const int wave = threadIdx.x >> 5;
  const float* x = xc + (size_t)b*P_SZ;
#pragma unroll
  for (int rep=0;rep<2;++rep){
    const int o = wave + 8*rep;                  // 0..15
    const float* Wrow = (o < 8) ? (Wi + (size_t)o*P_SZ) : (Wf + (size_t)(o-8)*P_SZ);
    float s = 0.f;
    for (int j=lane; j<P_SZ; j+=32) s += x[j]*Wrow[j];
#pragma unroll
    for (int off=16; off>0; off>>=1) s += __shfl_down(s, off);
    if (lane==0){
      if (o < 8) it[b*H_SZ+o]     = s + bi[o];
      else       ft[b*H_SZ+(o-8)] = s + bf[o-8];
    }
  }
}

// ---------------------------------------------------------------------------
// Fused state update + retrieval + GroupNorm.
// One block per (b,h), 128 threads (thread = row d).
// num[d] = f_e*(c0[d,:].q) + i_e*v[d]*(k.q)  -> single pass over c0.
// q/k/v staged to LDS via gfx1250 async-to-LDS (ASYNCcnt) when available.
// ---------------------------------------------------------------------------
__device__ __forceinline__ float block_reduce128(float v, float* red, int tid){
  red[tid] = v; __syncthreads();
  for (int s = 64; s > 0; s >>= 1){ if (tid < s) red[tid] += red[tid+s]; __syncthreads(); }
  const float r = red[0]; __syncthreads();
  return r;
}

__global__ __launch_bounds__(128)
void state_kernel(const float* __restrict__ q32, const float* __restrict__ k32,
                  const float* __restrict__ v32, const float* __restrict__ opre,
                  const float* __restrict__ it, const float* __restrict__ ft,
                  const float* __restrict__ c0, const float* __restrict__ n0,
                  const float* __restrict__ m0, const float* __restrict__ gng,
                  const float* __restrict__ gnb,
                  float* __restrict__ ct, float* __restrict__ nt,
                  float* __restrict__ mt, float* __restrict__ hn){
  __shared__ float sq[D_SZ], sk[D_SZ], sv[D_SZ], red[D_SZ];
  const int bh  = blockIdx.x;                 // b*H + h
  const int h   = bh & (H_SZ-1);
  const int tid = threadIdx.x;
  const size_t base = (size_t)bh * D_SZ;      // == b*HD + h*D

#if ASYNC_LDS
  __builtin_amdgcn_global_load_async_to_lds_b32(GPTR(q32 + base + tid), LPTR(&sq[tid]), 0, 0);
  __builtin_amdgcn_global_load_async_to_lds_b32(GPTR(k32 + base + tid), LPTR(&sk[tid]), 0, 0);
  __builtin_amdgcn_global_load_async_to_lds_b32(GPTR(v32 + base + tid), LPTR(&sv[tid]), 0, 0);
#else
  sq[tid] = q32[base+tid];
  sk[tid] = k32[base+tid];
  sv[tid] = v32[base+tid];
#endif

  const float iv = it[bh], fv = ft[bh], m0v = m0[bh];
  const float mtv = fmaxf(fv + m0v, iv);
  const float ie  = __expf(iv - mtv);
  const float fe  = __expf(fv - mtv + m0v);
  if (tid == 0) mt[bh] = mtv;

#if ASYNC_LDS
  __builtin_amdgcn_s_wait_asynccnt(0);
#endif
  __syncthreads();
  sk[tid] *= INV_SQRT_D;   // foreign reads of sk happen only after the barriers
                           // inside the block_reduce calls below

  const float kq  = block_reduce128(sk[tid]*sq[tid], red, tid);
  const float ntv = fe*n0[base+tid] + ie*sk[tid];
  nt[base+tid] = ntv;
  float den = block_reduce128(ntv*sq[tid], red, tid);
  den = fmaxf(den, 1.0f);

  const float4* c0r = (const float4*)(c0 + (base+tid)*(size_t)D_SZ);
  float4*       ctr = (float4*)      (ct + (base+tid)*(size_t)D_SZ);
  const float vd = sv[tid];
  float s0 = 0.f;
#pragma unroll 4
  for (int p4=0; p4<D_SZ/4; ++p4){
    const float4 cv = c0r[p4];
    const int p = p4*4;
    float4 o;
    o.x = fe*cv.x + ie*vd*sk[p+0];
    o.y = fe*cv.y + ie*vd*sk[p+1];
    o.z = fe*cv.z + ie*vd*sk[p+2];
    o.w = fe*cv.w + ie*vd*sk[p+3];
    ctr[p4] = o;
    s0 += cv.x*sq[p+0] + cv.y*sq[p+1] + cv.z*sq[p+2] + cv.w*sq[p+3];
  }
  const float num  = fe*s0 + ie*vd*kq;
  const float hval = sigmoidf_(opre[base+tid]) * (num/den);

  const float mu  = block_reduce128(hval, red, tid) * (1.0f/D_SZ);
  const float df  = hval - mu;
  const float var = block_reduce128(df*df, red, tid) * (1.0f/D_SZ);
  hn[base+tid] = df * rsqrtf(var + EPSV) * gng[h*D_SZ+tid] + gnb[h*D_SZ+tid];
}

// ---------------------------------------------------------------------------
// (hn + skip) * silu(r_t) -> f16 for the final down-projection GEMM
// ---------------------------------------------------------------------------
__global__ void epilogue1_kernel(const float* __restrict__ hn, const float* __restrict__ sk,
                                 const float* __restrict__ rt, h16* __restrict__ pre){
  const int g = blockIdx.x*blockDim.x + threadIdx.x;
  if (g >= B_SZ*HD_SZ) return;
  pre[g] = (h16)((hn[g] + sk[g]) * siluf_(rt[g]));
}

// ---------------------------------------------------------------------------
extern "C" void kernel_launch(void* const* d_in, const int* in_sizes, int n_in,
                              void* d_out, int out_size, void* d_ws, size_t ws_size,
                              hipStream_t stream) {
  (void)in_sizes; (void)n_in; (void)out_size; (void)ws_size;
  const float* seq   = (const float*)d_in[0];
  const float* c0    = (const float*)d_in[1];
  const float* n0    = (const float*)d_in[2];
  const float* m0    = (const float*)d_in[3];
  const float* ln_g  = (const float*)d_in[4];
  const float* ln_b  = (const float*)d_in[5];
  const float* gn_g  = (const float*)d_in[6];
  const float* gn_b  = (const float*)d_in[7];
  const float* W_upl = (const float*)d_in[8];
  const float* b_upl = (const float*)d_in[9];
  const float* W_upr = (const float*)d_in[10];
  const float* b_upr = (const float*)d_in[11];
  const float* W_down= (const float*)d_in[12];
  const float* b_down= (const float*)d_in[13];
  const float* conv_w= (const float*)d_in[14];
  const float* conv_b= (const float*)d_in[15];
  const float* W_skip= (const float*)d_in[16];
  const float* W_i   = (const float*)d_in[17];
  const float* b_i   = (const float*)d_in[18];
  const float* W_f   = (const float*)d_in[19];
  const float* b_f   = (const float*)d_in[20];
  const float* W_o   = (const float*)d_in[21];
  const float* b_o   = (const float*)d_in[22];
  const float* W_q   = (const float*)d_in[23];
  const float* b_q   = (const float*)d_in[24];
  const float* W_k   = (const float*)d_in[25];
  const float* b_k   = (const float*)d_in[26];
  const float* W_v   = (const float*)d_in[27];
  const float* b_v   = (const float*)d_in[28];

  // Output tuple, concatenated flat: out, c_t, n_t, m_t
  float* out = (float*)d_out;
  float* ct  = out + (size_t)B_SZ*I_SZ;
  float* nt  = ct  + (size_t)B_SZ*H_SZ*D_SZ*D_SZ;
  float* mt  = nt  + (size_t)B_SZ*H_SZ*D_SZ;

  // Workspace carve-up (256B aligned slabs)
  char* wsp = (char*)d_ws; size_t off = 0;
  auto alloc = [&](size_t bytes)->void*{
    void* p = wsp + off; off += (bytes + 255) & ~(size_t)255; return p; };

  h16* w16_upl  = (h16*)alloc((size_t)P_SZ *I_SZ*sizeof(h16));
  h16* w16_upr  = (h16*)alloc((size_t)HD_SZ*I_SZ*sizeof(h16));
  h16* w16_q    = (h16*)alloc((size_t)HD_SZ*P_SZ*sizeof(h16));
  h16* w16_k    = (h16*)alloc((size_t)HD_SZ*P_SZ*sizeof(h16));
  h16* w16_v    = (h16*)alloc((size_t)HD_SZ*P_SZ*sizeof(h16));
  h16* w16_o    = (h16*)alloc((size_t)HD_SZ*P_SZ*sizeof(h16));
  h16* w16_skip = (h16*)alloc((size_t)HD_SZ*P_SZ*sizeof(h16));
  h16* w16_down = (h16*)alloc((size_t)I_SZ *HD_SZ*sizeof(h16));
  h16* xn16     = (h16*)alloc((size_t)B_SZ*I_SZ*sizeof(h16));
  float* xt32   = (float*)alloc((size_t)B_SZ*P_SZ*sizeof(float));
  float* rt32   = (float*)alloc((size_t)B_SZ*HD_SZ*sizeof(float));
  float* xc32   = (float*)alloc((size_t)B_SZ*P_SZ*sizeof(float));
  h16* xt16     = (h16*)alloc((size_t)B_SZ*P_SZ*sizeof(h16));
  h16* xc16     = (h16*)alloc((size_t)B_SZ*P_SZ*sizeof(h16));
  float* q32    = (float*)alloc((size_t)B_SZ*HD_SZ*sizeof(float));
  float* k32    = (float*)alloc((size_t)B_SZ*HD_SZ*sizeof(float));
  float* v32    = (float*)alloc((size_t)B_SZ*HD_SZ*sizeof(float));
  float* opre32 = (float*)alloc((size_t)B_SZ*HD_SZ*sizeof(float));
  float* it32   = (float*)alloc((size_t)B_SZ*H_SZ*sizeof(float));
  float* ft32   = (float*)alloc((size_t)B_SZ*H_SZ*sizeof(float));
  float* hn32   = (float*)alloc((size_t)B_SZ*HD_SZ*sizeof(float));
  float* skip32 = (float*)alloc((size_t)B_SZ*HD_SZ*sizeof(float));
  h16* pre16    = (h16*)alloc((size_t)B_SZ*HD_SZ*sizeof(h16));

  // 1) weight conversions f32 -> f16
  f32_to_h16_kernel<<<2048, 256, 0, stream>>>(W_upl,  w16_upl,  P_SZ*I_SZ);
  f32_to_h16_kernel<<<1024, 256, 0, stream>>>(W_upr,  w16_upr,  HD_SZ*I_SZ);
  f32_to_h16_kernel<<<2048, 256, 0, stream>>>(W_q,    w16_q,    HD_SZ*P_SZ);
  f32_to_h16_kernel<<<2048, 256, 0, stream>>>(W_k,    w16_k,    HD_SZ*P_SZ);
  f32_to_h16_kernel<<<2048, 256, 0, stream>>>(W_v,    w16_v,    HD_SZ*P_SZ);
  f32_to_h16_kernel<<<2048, 256, 0, stream>>>(W_o,    w16_o,    HD_SZ*P_SZ);
  f32_to_h16_kernel<<<2048, 256, 0, stream>>>(W_skip, w16_skip, HD_SZ*P_SZ);
  f32_to_h16_kernel<<<1024, 256, 0, stream>>>(W_down, w16_down, I_SZ*HD_SZ);

  // 2) LayerNorm -> xn16
  layernorm_kernel<<<B_SZ, 256, 0, stream>>>(seq, ln_g, ln_b, xn16);

  // 3) up-projections (single-wave blocks: grid = N/32 x M/32)
  wmma_gemm_kernel<<<dim3(P_SZ/32,  B_SZ/32), 32, 0, stream>>>(xn16, w16_upl, b_upl, nullptr, xt32, B_SZ, P_SZ,  I_SZ);
  wmma_gemm_kernel<<<dim3(HD_SZ/32, B_SZ/32), 32, 0, stream>>>(xn16, w16_upr, b_upr, nullptr, rt32, B_SZ, HD_SZ, I_SZ);

  // 4) causal conv + SiLU, plus f16 casts of x_t and x_c
  conv_silu_kernel<<<(B_SZ*P_SZ+255)/256, 256, 0, stream>>>(xt32, conv_w, conv_b, xc32, xc16, xt16);

  // 5) q/k from x_c, v/o from x_t
  wmma_gemm_kernel<<<dim3(HD_SZ/32, B_SZ/32), 32, 0, stream>>>(xc16, w16_q, b_q, nullptr, q32,    B_SZ, HD_SZ, P_SZ);
  wmma_gemm_kernel<<<dim3(HD_SZ/32, B_SZ/32), 32, 0, stream>>>(xc16, w16_k, b_k, nullptr, k32,    B_SZ, HD_SZ, P_SZ);
  wmma_gemm_kernel<<<dim3(HD_SZ/32, B_SZ/32), 32, 0, stream>>>(xt16, w16_v, b_v, nullptr, v32,    B_SZ, HD_SZ, P_SZ);
  wmma_gemm_kernel<<<dim3(HD_SZ/32, B_SZ/32), 32, 0, stream>>>(xt16, w16_o, b_o, nullptr, opre32, B_SZ, HD_SZ, P_SZ);

  // 6) i/f gate projections (N=8 each)
  ifgate_kernel<<<B_SZ, 256, 0, stream>>>(xc32, W_i, b_i, W_f, b_f, it32, ft32);

  // 7) fused state update + retrieval + GroupNorm (writes c_t, n_t, m_t)
  state_kernel<<<B_SZ*H_SZ, 128, 0, stream>>>(q32, k32, v32, opre32, it32, ft32,
                                              c0, n0, m0, gn_g, gn_b,
                                              ct, nt, mt, hn32);

  // 8) skip projection, epilogue, down projection (+seq residual) -> out
  wmma_gemm_kernel<<<dim3(HD_SZ/32, B_SZ/32), 32, 0, stream>>>(xc16, w16_skip, nullptr, nullptr, skip32, B_SZ, HD_SZ, P_SZ);
  epilogue1_kernel<<<(B_SZ*HD_SZ+255)/256, 256, 0, stream>>>(hn32, skip32, rt32, pre16);
  wmma_gemm_kernel<<<dim3(I_SZ/32, B_SZ/32), 32, 0, stream>>>(pre16, w16_down, b_down, seq, out, B_SZ, I_SZ, HD_SZ);
}
